// TtMoeLayer_21388937134657
// MI455X (gfx1250) — matrix operand order repaired and checked by reading, not compiled, not run on hardware
//
#include <hip/hip_runtime.h>

// Mixtral-8x7B decode MoE layer, dense-over-experts (matches reference math):
//   logits = x @ gate_w ; top-2 softmax weights
//   act[e,b,f] = w[b,e] * silu(x@w1[e]) * (x@w3[e])     (bf16, in workspace)
//   out[b,h]   = sum_e act[e,b,:] @ w2[e]               (f32 partials -> bf16)
//
// Roofline: weights = 2.82 GB read once -> 121 us at 23.3 TB/s; bf16 WMMA
// compute (90 GFLOP) is ~5x cheaper -> HBM-bound weight streaming.
// ffn1: register-double-buffered weight tiles (2 K-steps of HBM traffic in
//       flight per wave) + LDS transpose staging + v_wmma_f32_16x16x32_bf16.
// ffn2: Tensor Data Mover (tensor_load_to_lds) streams the strided 32x16
//       weight tiles into double-buffered LDS, paced by s_wait_tensorcnt.

typedef unsigned short u16;
typedef __attribute__((ext_vector_type(16))) __bf16 v16bf;
typedef __attribute__((ext_vector_type(8)))  float  v8f;
typedef __attribute__((ext_vector_type(4)))  unsigned int v4u;
typedef __attribute__((ext_vector_type(8)))  int v8i;
typedef __attribute__((ext_vector_type(4)))  int v4i;

#define E_  8
#define B_  32
#define H_  4096
#define F_  14336
#define EF_ (E_ * F_)   // 114688
#define TPAD 40         // padded row stride (halves) of transposed LDS tile: 80B, 16B-aligned

__device__ __forceinline__ u16 f2bf(float f) {          // RNE f32 -> bf16
  unsigned u = __float_as_uint(f);
  return (u16)((u + 0x7FFFu + ((u >> 16) & 1u)) >> 16);
}
__device__ __forceinline__ float bf2f(u16 h) {
  return __uint_as_float(((unsigned)h) << 16);
}

// ---- WMMA fragment helpers -------------------------------------------------
// A (16x32 bf16, MxK) per-lane layout: lane L holds row M=L%16,
// one contiguous 32B run at K = (L/16)*16 .. +15 (matches B fragments).
__device__ __forceinline__ v16bf load_a_frag(const u16* base, size_t stride,
                                             int lane, int mbase) {
  const int m = mbase + (lane & 15);
  const int half = lane >> 4;
  const u16* p = base + (size_t)m * stride + half * 16;
  union { v16bf v; uint4 q[2]; } u;
  u.q[0] = *reinterpret_cast<const uint4*>(p);
  u.q[1] = *reinterpret_cast<const uint4*>(p + 8);
  return u.v;
}

// Staged 32x16 bf16 weight tile: lane L holds row k=L (16 halves, two b128
// global loads). Kept in registers one K-step ahead of consumption.
struct TileRegs { union { uint4 q[2]; u16 h[16]; } u; };

__device__ __forceinline__ void load_tile_regs(const u16* src, size_t stride,
                                               int lane, TileRegs& t) {
  const uint4* p = reinterpret_cast<const uint4*>(src + (size_t)lane * stride);
  t.u.q[0] = p[0];
  t.u.q[1] = p[1];
}

// Scatter the staged tile into wave-private LDS transposed: lds[n*TPAD + k].
__device__ __forceinline__ void store_tile_lds(u16* tile, int lane,
                                               const TileRegs& t) {
#pragma unroll
  for (int n = 0; n < 16; ++n) tile[n * TPAD + lane] = t.u.h[n];
}

// B (32x16 bf16, KxN) per-lane layout: lane L = column N=L%16, contiguous run
// K = (L/16)*16 .. +15  -> 32B at lds[(L%16)*TPAD + (L/16)*16] (two ds_load_b128).
__device__ __forceinline__ v16bf read_b_frag(const u16* tile, int lane) {
  const u16* p = tile + (size_t)(lane & 15) * TPAD + (lane >> 4) * 16;
  union { v16bf v; uint4 q[2]; } u;
  u.q[0] = *reinterpret_cast<const uint4*>(p);
  u.q[1] = *reinterpret_cast<const uint4*>(p + 8);
  return u.v;
}

// B fragment out of a row-major 32x16 LDS tile (as deposited by the TDM):
// element (k,n) at tile[k*16 + n]; lane L needs n=L%16, k=(L/16)*16..+15.
// volatile: the compiler never sees the TDM write this LDS, don't let it fold.
__device__ __forceinline__ v16bf read_b_frag_rowmajor(const volatile u16* tile,
                                                      int lane) {
  const int n = lane & 15;
  const int ks = (lane >> 4) * 16;
  union { v16bf v; u16 h[16]; } u;
#pragma unroll
  for (int j = 0; j < 16; ++j) u.h[j] = tile[(ks + j) * 16 + n];
  return u.v;
}

__device__ __forceinline__ v8f wmma_bf16(v16bf a, v16bf b, v8f c) {
  return __builtin_amdgcn_wmma_f32_16x16x32_bf16(
      /*neg_a=*/false, a, /*neg_b=*/false, b,
      /*c_mod=*/(short)0, c, /*reuse_a=*/false, /*reuse_b=*/false);
}

__device__ __forceinline__ float silu_fast(float h) {
  return h * __builtin_amdgcn_rcpf(1.f + __expf(-h));   // bf16-precision output
}

// ---- Tensor Data Mover: async 2D tile (32 rows x 16 cols bf16, row stride
// `stride` elements) from global to LDS, row-major. D# per ISA 8.3/8.4.
// This toolchain's builtin takes 6 args: (v4u, v8i, v4i, v4i, v8i, i32 cpol).
__device__ __forceinline__ void tdm_load_tile(const u16* gsrc, unsigned lds_off,
                                              unsigned stride) {
  const unsigned long long ga = (unsigned long long)(const void*)gsrc;
  v4u g0;
  g0.x = 1u;                                        // count=1, no gather
  g0.y = lds_off;                                   // lds_addr (bytes)
  g0.z = (unsigned)(ga & 0xFFFFFFFFu);              // global_addr[31:0]
  g0.w = (unsigned)((ga >> 32) & 0x1FFFFFFu) |      // global_addr[56:32]
         (2u << 30);                                // type=2 ("image")
  v8i g1;
  g1[0] = (int)(1u << 16);        // workgroup_mask=0 | data_size=1 (2 bytes)
  g1[1] = (int)(16u << 16);       // atomic_barrier_addr=0 | tensor_dim0=16 lo
  g1[2] = (int)(32u << 16);       // tensor_dim0 hi=0 | tensor_dim1=32 lo
  g1[3] = (int)(16u << 16);       // tensor_dim1 hi=0 | tile_dim0=16
  g1[4] = 32;                     // tile_dim1=32 | tile_dim2=0
  g1[5] = (int)stride;            // tensor_dim0_stride[31:0] (elements)
  g1[6] = 0;                      // stride0 hi | tensor_dim1_stride lo
  g1[7] = 0;                      // tensor_dim1_stride hi
  v4i g2 = {0, 0, 0, 0};          // 2D tile: dims 2..4 unused
  v4i g3 = {0, 0, 0, 0};
  v8i g4 = {0, 0, 0, 0, 0, 0, 0, 0};
  __builtin_amdgcn_tensor_load_to_lds(g0, g1, g2, g3, g4, /*cpol=*/0);
}

// ---- Kernel 1: gate + top-2 routing (tiny) ---------------------------------
__global__ __launch_bounds__(256) void moe_gate_route(const u16* __restrict__ x,
                                                      const u16* __restrict__ gw,
                                                      float* __restrict__ wgt) {
  __shared__ float lg[B_][E_];
  const int t = threadIdx.x;
  const int b = t >> 3, e = t & 7;
  float acc = 0.f;
  for (int h = 0; h < H_; ++h)
    acc += bf2f(x[b * H_ + h]) * bf2f(gw[h * E_ + e]);
  lg[b][e] = bf2f(f2bf(acc));   // round to bf16: tie-compares match reference
  __syncthreads();
  if (t < B_) {
    float l[E_];
#pragma unroll
    for (int i = 0; i < E_; ++i) l[i] = lg[t][i];
    float ex0 = l[0];
#pragma unroll
    for (int i = 1; i < E_; ++i) ex0 = fmaxf(ex0, l[i]);
    float ex1 = -__builtin_inff();            // masked max (all ties masked)
#pragma unroll
    for (int i = 0; i < E_; ++i) if (l[i] != ex0) ex1 = fmaxf(ex1, l[i]);
    const float wtop = 1.f / (1.f + __expf(ex1 - ex0));
#pragma unroll
    for (int i = 0; i < E_; ++i) {
      float w = (l[i] == ex0 ? wtop : 0.f) + (l[i] == ex1 ? (1.f - wtop) : 0.f);
      wgt[t * E_ + i] = bf2f(f2bf(w));
    }
  }
}

// ---- Kernel 2: stream w1/w3, act = wgt * silu(x@w1) * (x@w3) ---------------
// grid = (F/128, E), block = 256 (8 waves). Wave owns a 32x16 tile of F.
// Software-pipelined: weight tile for K-step i+1 lives in registers while
// K-step i is transposed through LDS and consumed by 4 WMMAs.
__global__ __launch_bounds__(256) void moe_ffn1(const u16* __restrict__ x,
                                                const u16* __restrict__ w1,
                                                const u16* __restrict__ w3,
                                                const float* __restrict__ wgt,
                                                u16* __restrict__ act) {
  __shared__ __align__(16) u16 lds[8 * 2 * 16 * TPAD];   // 20.5 KB
  const int lane = threadIdx.x & 31;
  const int wave = threadIdx.x >> 5;
  const int e = blockIdx.y;
  const int f0 = blockIdx.x * 128 + wave * 16;
  u16* t1 = &lds[(wave * 2 + 0) * 16 * TPAD];
  u16* t3 = &lds[(wave * 2 + 1) * 16 * TPAD];

  const size_t wbase = (size_t)e * H_ * F_ + f0;
  v8f hc0 = {}, hc1 = {}, gc0 = {}, gc1 = {};

  TileRegs r1, r3;                                       // prologue: K-step 0
  load_tile_regs(w1 + wbase, F_, lane, r1);
  load_tile_regs(w3 + wbase, F_, lane, r3);

  for (int kk = 0; kk < H_; kk += 32) {
    store_tile_lds(t1, lane, r1);
    store_tile_lds(t3, lane, r3);
    const int kn = kk + 32;
    if (kn < H_) {                 // issue next K-step's HBM loads immediately
      const u16* s1 = w1 + wbase + (size_t)kn * F_;
      const u16* s3 = w3 + wbase + (size_t)kn * F_;
      load_tile_regs(s1, F_, lane, r1);
      load_tile_regs(s3, F_, lane, r3);
      __builtin_prefetch((const void*)(s1 + (size_t)(32 + lane) * F_), 0, 1);
      __builtin_prefetch((const void*)(s3 + (size_t)(32 + lane) * F_), 0, 1);
    }
    v16bf a0 = load_a_frag(x + kk, H_, lane, 0);         // x: L2-resident
    v16bf a1 = load_a_frag(x + kk, H_, lane, 16);
    asm volatile("s_wait_dscnt 0" ::: "memory");         // LDS transpose done
    v16bf b1 = read_b_frag(t1, lane);
    v16bf b3 = read_b_frag(t3, lane);
    hc0 = wmma_bf16(a0, b1, hc0);
    hc1 = wmma_bf16(a1, b1, hc1);
    gc0 = wmma_bf16(a0, b3, gc0);
    gc1 = wmma_bf16(a1, b3, gc1);
  }

  // Epilogue: silu(h)*g, fold routing weight, store bf16.
  // C layout: lanes 0-15 rows M=r, lanes 16-31 rows M=8+r; N = lane%16.
  const int n = lane & 15;
  const int rg = (lane >> 4) * 8;
#pragma unroll
  for (int r = 0; r < 8; ++r) {
    const int m0 = rg + r, m1 = 16 + rg + r;
    const float v0 = silu_fast(hc0[r]) * gc0[r] * wgt[m0 * E_ + e];
    const float v1 = silu_fast(hc1[r]) * gc1[r] * wgt[m1 * E_ + e];
    act[(size_t)m0 * EF_ + (size_t)e * F_ + f0 + n] = f2bf(v0);
    act[(size_t)m1 * EF_ + (size_t)e * F_ + f0 + n] = f2bf(v1);
  }
}

// ---- Kernel 3: stream w2 via the Tensor Data Mover -------------------------
// grid = (H/128, E), block = 256. Per wave: TDM pulls the strided 32x16 w2
// tile into double-buffered LDS (TENSORcnt-paced, 2 tiles in flight); the
// wave only issues A-fragment loads and WMMAs.
__global__ __launch_bounds__(256) void moe_ffn2(const u16* __restrict__ act,
                                                const u16* __restrict__ w2,
                                                float* __restrict__ partial) {
  __shared__ __align__(16) u16 lds2[8 * 2 * 512];        // 16 KB, row-major tiles
  const int lane = threadIdx.x & 31;
  const int wave = threadIdx.x >> 5;
  const int e = blockIdx.y;
  const int h0 = blockIdx.x * 128 + wave * 16;

  volatile u16* buf[2];
  buf[0] = &lds2[wave * 2 * 512];
  buf[1] = &lds2[wave * 2 * 512 + 512];
  const unsigned off0 = (unsigned)(wave * 2 * 512 * sizeof(u16));
  const unsigned off1 = off0 + 1024;

  const u16* arow = act + (size_t)e * F_;                // + m*EF_ + kk
  const u16* wrow = w2 + (size_t)e * F_ * H_ + h0;       // + (kk)*H_
  v8f c0 = {}, c1 = {};

  tdm_load_tile(wrow, off0, H_);                         // prologue: tile 0

  for (int kk = 0; kk < F_; kk += 32) {
    const int i = kk >> 5;
    const int kn = kk + 32;
    if (kn < F_) {                                       // keep 2 tiles in flight
      tdm_load_tile(wrow + (size_t)kn * H_, ((i + 1) & 1) ? off1 : off0, H_);
      __builtin_amdgcn_s_wait_tensorcnt(1);              // tile i landed
    } else {
      __builtin_amdgcn_s_wait_tensorcnt(0);
    }
    v16bf a0 = load_a_frag(arow + kk, EF_, lane, 0);
    v16bf a1 = load_a_frag(arow + kk, EF_, lane, 16);
    v16bf bb = read_b_frag_rowmajor(buf[i & 1], lane);
    c0 = wmma_bf16(a0, bb, c0);
    c1 = wmma_bf16(a1, bb, c1);
  }

  const int n = lane & 15;
  const int rg = (lane >> 4) * 8;
#pragma unroll
  for (int r = 0; r < 8; ++r) {
    partial[((size_t)e * B_ + rg + r) * H_ + h0 + n] = c0[r];
    partial[((size_t)e * B_ + 16 + rg + r) * H_ + h0 + n] = c1[r];
  }
}

// ---- Kernel 4: deterministic reduce over experts, f32 -> bf16 --------------
__global__ __launch_bounds__(256) void moe_reduce(const float* __restrict__ partial,
                                                  u16* __restrict__ out) {
  const int idx = blockIdx.x * 256 + threadIdx.x;        // < B_*H_
  float s = 0.f;
#pragma unroll
  for (int e = 0; e < E_; ++e)
    s += partial[(size_t)e * B_ * H_ + idx];
  out[idx] = f2bf(s);
}

extern "C" void kernel_launch(void* const* d_in, const int* in_sizes, int n_in,
                              void* d_out, int out_size, void* d_ws, size_t ws_size,
                              hipStream_t stream) {
  const u16* x  = (const u16*)d_in[0];   // (1,1,32,4096) bf16
  const u16* gw = (const u16*)d_in[1];   // (4096,8) bf16
  const u16* w1 = (const u16*)d_in[2];   // (8,4096,14336) bf16
  const u16* w3 = (const u16*)d_in[3];   // (8,4096,14336) bf16
  const u16* w2 = (const u16*)d_in[4];   // (8,14336,4096) bf16
  u16* out = (u16*)d_out;                // (1,1,32,4096) bf16

  // workspace layout: [weights f32 1KB][act bf16 7.34MB][partials f32 4MB]
  char* ws = (char*)d_ws;
  float* wgt     = (float*)ws;
  u16*   act     = (u16*)(ws + 1024);
  float* partial = (float*)(ws + 1024 + (size_t)B_ * EF_ * sizeof(u16));

  moe_gate_route<<<1, 256, 0, stream>>>(x, gw, wgt);
  moe_ffn1<<<dim3(F_ / 128, E_), 256, 0, stream>>>(x, w1, w3, wgt, act);
  moe_ffn2<<<dim3(H_ / 128, E_), 256, 0, stream>>>(act, w2, partial);
  moe_reduce<<<(B_ * H_) / 256, 256, 0, stream>>>(partial, out);
}
